// XASGNN_88356067213451
// MI455X (gfx1250) — matrix operand order repaired and checked by reading, not compiled, not run on hardware
//
#include <hip/hip_runtime.h>

typedef __attribute__((ext_vector_type(16))) _Float16 v16h;
typedef __attribute__((ext_vector_type(8)))  _Float16 v8h;
typedef __attribute__((ext_vector_type(8)))  float    v8f;

__device__ __constant__ float c_roots[9] = {
  3.14159265358979f, 6.28318530717959f, 9.42477796076938f,
  4.49340945790906f, 7.72525183693771f, 10.90412165942890f,
  5.76345919689455f, 9.09501133047625f, 12.32294097056454f };

__device__ inline float d_silu(float x){ return x / (1.f + __expf(-x)); }
__device__ inline float d_sigm(float x){ return 1.f / (1.f + __expf(-x)); }

// spherical Bessel j0..j3 (renamed to avoid HIP math header j0f/j1f clash)
__device__ inline float sbj0(float x){ return sinf(x)/x; }
__device__ inline float sbj1(float x){ float s=sinf(x), c=cosf(x); return s/(x*x) - c/x; }
__device__ inline float sbj2(float x){ float s=sinf(x), c=cosf(x); float x2=x*x;
  return (3.f/(x2*x) - 1.f/x)*s - 3.f*c/x2; }
__device__ inline float sbj3(float x){ float s=sinf(x), c=cosf(x); float x2=x*x,x3=x2*x,x4=x2*x2;
  return (15.f/x4 - 6.f/x2)*s - (15.f/x3 - 1.f/x)*c; }

__device__ inline v8f vzero8(){ v8f z = {0.f,0.f,0.f,0.f,0.f,0.f,0.f,0.f}; return z; }

// wave-local LDS ordering: same-wave DS ops are in-order; just stop compiler
// reordering and drain DScnt.
__device__ inline void lds_fence(){
  __builtin_amdgcn_wave_barrier();
  asm volatile("s_wait_dscnt 0" ::: "memory");
}

// A fragment (16x32 f16, M=row across lanes 0..15): lane<16 -> K {0..15},
// lane>=16 -> K {16..31}; two contiguous 16B chunks from row-major LDS.
__device__ inline v16h frag_a(const _Float16* B, int stride, int kt, int mrow, int kofs){
  const _Float16* p = B + mrow*stride + kt*32 + kofs;
  v8h lo = *(const v8h*)p;
  v8h hi = *(const v8h*)(p + 8);
  v16h r;
#pragma unroll
  for (int i=0;i<8;++i){ r[i]=lo[i]; r[i+8]=hi[i]; }
  return r;
}

// B fragment (32x16 f16): weights stored transposed [64 cols][K rows] so each
// lane's 16 K-values (k = kt*32 + (lane&16?16:0) + 0..15) are contiguous.
__device__ inline v16h frag_b(const _Float16* W, int K, int kt, int nt, int ncol, int kofs){
  const _Float16* p = W + (nt*16 + ncol)*K + kt*32 + kofs;
  return *(const v16h*)p;
}

#define WMMA_F16(a,b,c) __builtin_amdgcn_wmma_f32_16x16x32_f16(false,(a),false,(b),(short)0,(c),false,false)

// ---------------------------------------------------------------------------
// gated-MLP stack over 16-edge tiles, one wave per tile, 2 waves/block.
// mode 0: gate branch (silu,silu,sigmoid) -> gbuf
// mode 1: value branch (silu x3) * gbuf * (rbf@ewW) -> edgeF +=
// mode 2: same product -> atomicAdd into nodeOut[edge_dst]
// ---------------------------------------------------------------------------
__global__ __launch_bounds__(64) void k_gated_wmma(
    const float* __restrict__ nodeF, float* __restrict__ edgeF,
    float* __restrict__ nodeOut, float* __restrict__ gbuf,
    const int* __restrict__ esrc, const int* __restrict__ edst,
    const _Float16* __restrict__ W0t, const _Float16* __restrict__ W1t,
    const _Float16* __restrict__ W2t,
    const float* __restrict__ b0, const float* __restrict__ b1,
    const float* __restrict__ b2,
    const float* __restrict__ ewW, const float* __restrict__ rbf,
    int nEdges, int mode)
{
  __shared__ __align__(32) _Float16 sW0[64*192];
  __shared__ __align__(32) _Float16 sW1[64*64];
  __shared__ __align__(32) _Float16 sW2[64*64];
  __shared__ float sBias[3*64];
  __shared__ float sEw[9*64];
  __shared__ __align__(32) _Float16 sM[2][16*192];
  __shared__ __align__(32) _Float16 sT[2][16*64];
  __shared__ float sRbf[2][16*9];

  const int tid = threadIdx.x;
  for (int i=tid;i<64*192;i+=64) sW0[i]=W0t[i];
  for (int i=tid;i<64*64;i+=64){ sW1[i]=W1t[i]; sW2[i]=W2t[i]; }
  if (tid < 64){ sBias[tid]=b0[tid]; sBias[64+tid]=b1[tid]; sBias[128+tid]=b2[tid]; }
  for (int i=tid;i<9*64;i+=64) sEw[i]=ewW[i];
  __syncthreads();

  const int wave = tid >> 5, lane = tid & 31;
  _Float16* M = sM[wave];
  _Float16* T = sT[wave];
  float*    R = sRbf[wave];
  const int mrow  = lane & 15;
  const int ncol  = lane & 15;
  const int kA    = (lane & 16) ? 16 : 0;
  const int kB    = (lane & 16) ? 16 : 0;
  const int rbase = (lane & 16) ? 8  : 0;

  const int nTiles = nEdges >> 4;
  for (int tile = blockIdx.x*2 + wave; tile < nTiles; tile += gridDim.x*2) {
    const int base = tile << 4;
    // ---- stage m = [vi | vj | edge_feat] as f16 (float4 granularity) ----
    for (int idx = lane; idx < 16*48; idx += 32) {
      int row = idx / 48, c4 = idx % 48;
      int e = base + row;
      const float* p;
      if (c4 < 16)      p = nodeF + (size_t)esrc[e]*64 + c4*4;
      else if (c4 < 32) p = nodeF + (size_t)edst[e]*64 + (c4-16)*4;
      else              p = edgeF + (size_t)e*64 + (c4-32)*4;
      float4 v = *(const float4*)p;
      _Float16* q = M + row*192 + c4*4;
      q[0]=(_Float16)v.x; q[1]=(_Float16)v.y; q[2]=(_Float16)v.z; q[3]=(_Float16)v.w;
    }
    if (mode != 0)
      for (int idx = lane; idx < 16*9; idx += 32) R[idx] = rbf[(size_t)base*9 + idx];
    lds_fence();

    // ---- layer 0: [16,192] x [192,64] ----
    v8f acc[4];
#pragma unroll
    for (int nt=0; nt<4; ++nt) acc[nt] = vzero8();
#pragma unroll
    for (int kt=0; kt<6; ++kt) {
      v16h a = frag_a(M, 192, kt, mrow, kA);
#pragma unroll
      for (int nt=0; nt<4; ++nt)
        acc[nt] = WMMA_F16(a, frag_b(sW0, 192, kt, nt, ncol, kB), acc[nt]);
    }
#pragma unroll
    for (int nt=0; nt<4; ++nt) {
      int col = nt*16 + ncol; float bb = sBias[col];
#pragma unroll
      for (int r=0;r<8;++r)
        T[(rbase+r)*64 + col] = (_Float16)d_silu(acc[nt][r] + bb);
    }
    lds_fence();

    // ---- layer 1: [16,64] x [64,64] ----
#pragma unroll
    for (int nt=0; nt<4; ++nt) acc[nt] = vzero8();
#pragma unroll
    for (int kt=0; kt<2; ++kt) {
      v16h a = frag_a(T, 64, kt, mrow, kA);
#pragma unroll
      for (int nt=0; nt<4; ++nt)
        acc[nt] = WMMA_F16(a, frag_b(sW1, 64, kt, nt, ncol, kB), acc[nt]);
    }
    lds_fence();   // all reads of T done before overwrite
#pragma unroll
    for (int nt=0; nt<4; ++nt) {
      int col = nt*16 + ncol; float bb = sBias[64 + col];
#pragma unroll
      for (int r=0;r<8;++r)
        T[(rbase+r)*64 + col] = (_Float16)d_silu(acc[nt][r] + bb);
    }
    lds_fence();

    // ---- layer 2: [16,64] x [64,64] ----
#pragma unroll
    for (int nt=0; nt<4; ++nt) acc[nt] = vzero8();
#pragma unroll
    for (int kt=0; kt<2; ++kt) {
      v16h a = frag_a(T, 64, kt, mrow, kA);
#pragma unroll
      for (int nt=0; nt<4; ++nt)
        acc[nt] = WMMA_F16(a, frag_b(sW2, 64, kt, nt, ncol, kB), acc[nt]);
    }

    // ---- epilogue ----
#pragma unroll
    for (int nt=0; nt<4; ++nt) {
      int col = nt*16 + ncol; float bb = sBias[128 + col];
#pragma unroll
      for (int r=0;r<8;++r) {
        int row = rbase + r;
        size_t e = (size_t)(base + row);
        float v = acc[nt][r] + bb;
        if (mode == 0) {
          gbuf[e*64 + col] = d_sigm(v);
        } else {
          v = d_silu(v);
          float g = gbuf[e*64 + col];
          float ew = 0.f;
#pragma unroll
          for (int r9=0;r9<9;++r9) ew += R[row*9 + r9] * sEw[r9*64 + col];
          float o = v * g * ew;
          if (mode == 1) edgeF[e*64 + col] += o;
          else atomicAdd(nodeOut + (size_t)edst[base+row]*64 + col, o);
        }
      }
    }
    lds_fence();   // drain before next tile re-stages M
  }
}

// ---------------------------------------------------------------------------
// support kernels
// ---------------------------------------------------------------------------
__global__ void k_transpose_h(const float* __restrict__ in, _Float16* __restrict__ out,
                              int K, int total){
  int i = blockIdx.x*blockDim.x + threadIdx.x;
  if (i >= total) return;
  int g = i / (K*64); int rem = i % (K*64); int k = rem / 64; int c = rem % 64;
  out[((size_t)g*64 + c)*K + k] = (_Float16)in[i];
}

__global__ void k_node_init(const int* __restrict__ nt, const float* __restrict__ emb,
                            float* __restrict__ out, int total){
  int i = blockIdx.x*blockDim.x + threadIdx.x;
  if (i >= total) return;
  out[i] = emb[nt[i>>6]*64 + (i & 63)];
}

__global__ void k_edge_geom(const float* __restrict__ dist, const float* __restrict__ bvec,
                            float* __restrict__ rbf, float* __restrict__ unitv,
                            float* __restrict__ cutv, int E){
  int e = blockIdx.x*blockDim.x + threadIdx.x;
  if (e >= E) return;
  float r = dist[e];
  const float SQ = 0.17677669529663689f;  // sqrt(2/4^3)
#pragma unroll
  for (int n=0;n<3;++n){
    float z0=c_roots[n], z1=c_roots[3+n], z2=c_roots[6+n];
    rbf[(size_t)e*9 + n    ] = (SQ/fabsf(sbj1(z0))) * sbj0(r*z0*0.25f);
    rbf[(size_t)e*9 + 3 + n] = (SQ/fabsf(sbj2(z1))) * sbj1(r*z1*0.25f);
    rbf[(size_t)e*9 + 6 + n] = (SQ/fabsf(sbj3(z2))) * sbj2(r*z2*0.25f);
  }
  float x = r*0.25f, x2=x*x, x3=x2*x;
  cutv[e] = (r < 4.0f) ? (1.f - 6.f*x3*x2 + 15.f*x2*x2 - 10.f*x3) : 0.f;
  float vx=bvec[(size_t)e*3], vy=bvec[(size_t)e*3+1], vz=bvec[(size_t)e*3+2];
  float inv = rsqrtf(vx*vx + vy*vy + vz*vz);
  unitv[(size_t)e*3]=vx*inv; unitv[(size_t)e*3+1]=vy*inv; unitv[(size_t)e*3+2]=vz*inv;
}

__global__ void k_edge_emb(const float* __restrict__ rbf, const float* __restrict__ W,
                           const float* __restrict__ b, float* __restrict__ ef, int total){
  int i = blockIdx.x*blockDim.x + threadIdx.x;
  if (i >= total) return;
  int e = i >> 6, c = i & 63;
  float acc = b[c];
#pragma unroll
  for (int r=0;r<9;++r) acc += rbf[(size_t)e*9 + r] * W[r*64 + c];
  ef[i] = d_silu(acc);
}

__global__ void k_triplet(const int* __restrict__ ts, const int* __restrict__ td,
                          const int* __restrict__ edst, const float* __restrict__ rbf,
                          const float* __restrict__ unitv, const float* __restrict__ cutv,
                          float* __restrict__ tb3, float* __restrict__ tw,
                          int* __restrict__ endAtom, int T){
  int t = blockIdx.x*blockDim.x + threadIdx.x;
  if (t >= T) return;
  int s = ts[t], d = td[t];
  float cs = unitv[(size_t)s*3]*unitv[(size_t)d*3]
           + unitv[(size_t)s*3+1]*unitv[(size_t)d*3+1]
           + unitv[(size_t)s*3+2]*unitv[(size_t)d*3+2];
  cs = fminf(1.f, fmaxf(-1.f, cs));
  float p0 = 0.28209479177387814f;
  float p1 = 0.48860251190291992f * cs;
  float p2 = 0.63078313050504001f * 0.5f * (3.f*cs*cs - 1.f);
#pragma unroll
  for (int n=0;n<3;++n){
    tb3[(size_t)t*9 + n    ] = rbf[(size_t)d*9 + n    ] * p0;
    tb3[(size_t)t*9 + 3 + n] = rbf[(size_t)d*9 + 3 + n] * p1;
    tb3[(size_t)t*9 + 6 + n] = rbf[(size_t)d*9 + 6 + n] * p2;
  }
  tw[t] = cutv[s] * cutv[d];
  endAtom[t] = edst[d];
}

__global__ void k_atoms(const float* __restrict__ nodeF, const float* __restrict__ W,
                        const float* __restrict__ b, float* __restrict__ atoms, int N){
  int n = blockIdx.x*blockDim.x + threadIdx.x;
  if (n >= N) return;
  float acc[9];
#pragma unroll
  for (int j=0;j<9;++j) acc[j] = b[j];
  for (int k=0;k<64;++k){
    float f = nodeF[(size_t)n*64 + k];
#pragma unroll
    for (int j=0;j<9;++j) acc[j] += f * W[k*9 + j];
  }
#pragma unroll
  for (int j=0;j<9;++j) atoms[(size_t)n*9 + j] = d_sigm(acc[j]);
}

__global__ void k_trip_scatter(const float* __restrict__ tb3, const float* __restrict__ atoms,
                               const float* __restrict__ tw, const int* __restrict__ endAtom,
                               const int* __restrict__ ts, float* __restrict__ nb9, int T){
  int t = blockIdx.x*blockDim.x + threadIdx.x;
  if (t >= T) return;
  int e = ts[t]; int ea = endAtom[t]; float w = tw[t];
#pragma unroll
  for (int j=0;j<9;++j)
    atomicAdd(&nb9[(size_t)e*9 + j], tb3[(size_t)t*9 + j] * atoms[(size_t)ea*9 + j] * w);
}

__global__ void k_edge_tb(const float* __restrict__ nb9, const float* __restrict__ W,
                          float* __restrict__ ef, int total){
  int i = blockIdx.x*blockDim.x + threadIdx.x;
  if (i >= total) return;
  int e = i >> 6, c = i & 63;
  float u = 0.f, v = 0.f;
#pragma unroll
  for (int r=0;r<9;++r){
    float x = nb9[(size_t)e*9 + r];
    u += x * W[r*64 + c];
    v += x * W[576 + r*64 + c];
  }
  ef[i] += d_silu(u) * d_sigm(v);
}

__global__ void k_gather(const float* __restrict__ nodeF, const int* __restrict__ idx,
                         float* __restrict__ out, int total){
  int i = blockIdx.x*blockDim.x + threadIdx.x;
  if (i >= total) return;
  out[i] = nodeF[(size_t)idx[i>>6]*64 + (i & 63)];
}

// ---------------------------------------------------------------------------
extern "C" void kernel_launch(void* const* d_in, const int* in_sizes, int n_in,
                              void* d_out, int out_size, void* d_ws, size_t ws_size,
                              hipStream_t stream) {
  const int*   node_type = (const int*)d_in[0];
  const int*   edge_src  = (const int*)d_in[1];
  const int*   edge_dst  = (const int*)d_in[2];
  const int*   trip_src  = (const int*)d_in[3];
  const int*   trip_dst  = (const int*)d_in[4];
  const int*   absorber  = (const int*)d_in[5];
  const float* bond_dist = (const float*)d_in[6];
  const float* bond_vec  = (const float*)d_in[7];
  const float* node_embed= (const float*)d_in[8];
  const float* edge_emb_W= (const float*)d_in[9];
  const float* edge_emb_b= (const float*)d_in[10];
  const float* tb_atom_W = (const float*)d_in[11];
  const float* tb_atom_b = (const float*)d_in[12];
  const float* tb_bond_W = (const float*)d_in[13];
  const float* eu_W0=(const float*)d_in[14]; const float* eu_b0=(const float*)d_in[15];
  const float* eu_W1=(const float*)d_in[16]; const float* eu_b1=(const float*)d_in[17];
  const float* eu_W2=(const float*)d_in[18]; const float* eu_b2=(const float*)d_in[19];
  const float* ew_W =(const float*)d_in[20];
  const float* nu_W0=(const float*)d_in[21]; const float* nu_b0=(const float*)d_in[22];
  const float* nu_W1=(const float*)d_in[23]; const float* nu_b1=(const float*)d_in[24];
  const float* nu_W2=(const float*)d_in[25]; const float* nu_b2=(const float*)d_in[26];
  const float* nw_W =(const float*)d_in[27];

  const int N = in_sizes[0];
  const int E = in_sizes[6];
  const int T = in_sizes[3];
  const int B = in_sizes[5];

  char* ws = (char*)d_ws; size_t off = 0;
  auto alloc = [&](size_t bytes)->char* {
    char* p = ws + off; off += (bytes + 255) & ~(size_t)255; return p;
  };
  float*     rbf    = (float*)alloc((size_t)E*9*4);
  float*     unitv  = (float*)alloc((size_t)E*3*4);
  float*     cutv   = (float*)alloc((size_t)E*4);
  float*     nodeA  = (float*)alloc((size_t)N*64*4);
  float*     nodeB  = (float*)alloc((size_t)N*64*4);
  float*     edgeF  = (float*)alloc((size_t)E*64*4);
  float*     tb3    = (float*)alloc((size_t)T*9*4);
  float*     tw     = (float*)alloc((size_t)T*4);
  int*       endAt  = (int*)  alloc((size_t)T*4);
  float*     atoms  = (float*)alloc((size_t)N*9*4);
  float*     nb9    = (float*)alloc((size_t)E*9*4);
  float*     gbuf   = (float*)alloc((size_t)E*64*4);
  _Float16*  euW0h  = (_Float16*)alloc((size_t)6*192*64*2);
  _Float16*  euW1h  = (_Float16*)alloc((size_t)6*64*64*2);
  _Float16*  euW2h  = (_Float16*)alloc((size_t)6*64*64*2);
  _Float16*  nuW0h  = (_Float16*)alloc((size_t)6*192*64*2);
  _Float16*  nuW1h  = (_Float16*)alloc((size_t)6*64*64*2);
  _Float16*  nuW2h  = (_Float16*)alloc((size_t)6*64*64*2);
  (void)ws_size; (void)n_in; (void)out_size;

  // f16 + transpose weight conversion ([G][K][64] -> [G][64][K])
  {
    int t0 = 6*192*64, t1 = 6*64*64;
    k_transpose_h<<<(t0+255)/256,256,0,stream>>>(eu_W0, euW0h, 192, t0);
    k_transpose_h<<<(t1+255)/256,256,0,stream>>>(eu_W1, euW1h,  64, t1);
    k_transpose_h<<<(t1+255)/256,256,0,stream>>>(eu_W2, euW2h,  64, t1);
    k_transpose_h<<<(t0+255)/256,256,0,stream>>>(nu_W0, nuW0h, 192, t0);
    k_transpose_h<<<(t1+255)/256,256,0,stream>>>(nu_W1, nuW1h,  64, t1);
    k_transpose_h<<<(t1+255)/256,256,0,stream>>>(nu_W2, nuW2h,  64, t1);
  }

  k_node_init<<<(N*64+255)/256,256,0,stream>>>(node_type, node_embed, nodeA, N*64);
  k_edge_geom<<<(E+255)/256,256,0,stream>>>(bond_dist, bond_vec, rbf, unitv, cutv, E);
  k_edge_emb<<<(E*64+255)/256,256,0,stream>>>(rbf, edge_emb_W, edge_emb_b, edgeF, E*64);
  k_triplet<<<(T+255)/256,256,0,stream>>>(trip_src, trip_dst, edge_dst, rbf, unitv, cutv,
                                          tb3, tw, endAt, T);

  float* cur = nodeA; float* nxt = nodeB;
  const int tiles = E >> 4;
  const int ggrid = (tiles + 1) / 2;      // 2 waves (tiles) per 64-thread block

  for (int i = 0; i < 3; ++i) {
    // three-body channel
    k_atoms<<<(N+127)/128,128,0,stream>>>(cur, tb_atom_W + (size_t)i*64*9,
                                          tb_atom_b + i*9, atoms, N);
    (void)hipMemsetAsync(nb9, 0, (size_t)E*9*4, stream);
    k_trip_scatter<<<(T+255)/256,256,0,stream>>>(tb3, atoms, tw, endAt, trip_src, nb9, T);
    k_edge_tb<<<(E*64+255)/256,256,0,stream>>>(nb9, tb_bond_W + (size_t)i*2*9*64, edgeF, E*64);

    // edge update: gate branch then fused value*gate*(rbf@ewW) accumulation
    k_gated_wmma<<<ggrid,64,0,stream>>>(cur, edgeF, nullptr, gbuf, edge_src, edge_dst,
        euW0h + (size_t)(i*2+1)*192*64, euW1h + (size_t)(i*2+1)*64*64,
        euW2h + (size_t)(i*2+1)*64*64,
        eu_b0 + (i*2+1)*64, eu_b1 + (i*2+1)*64, eu_b2 + (i*2+1)*64,
        ew_W + (size_t)i*9*64, rbf, E, 0);
    k_gated_wmma<<<ggrid,64,0,stream>>>(cur, edgeF, nullptr, gbuf, edge_src, edge_dst,
        euW0h + (size_t)(i*2+0)*192*64, euW1h + (size_t)(i*2+0)*64*64,
        euW2h + (size_t)(i*2+0)*64*64,
        eu_b0 + (i*2+0)*64, eu_b1 + (i*2+0)*64, eu_b2 + (i*2+0)*64,
        ew_W + (size_t)i*9*64, rbf, E, 1);

    // node update: reads pre-update node feats (cur), accumulates into copy (nxt)
    k_gated_wmma<<<ggrid,64,0,stream>>>(cur, edgeF, nullptr, gbuf, edge_src, edge_dst,
        nuW0h + (size_t)(i*2+1)*192*64, nuW1h + (size_t)(i*2+1)*64*64,
        nuW2h + (size_t)(i*2+1)*64*64,
        nu_b0 + (i*2+1)*64, nu_b1 + (i*2+1)*64, nu_b2 + (i*2+1)*64,
        nw_W + (size_t)i*9*64, rbf, E, 0);
    (void)hipMemcpyAsync(nxt, cur, (size_t)N*64*4, hipMemcpyDeviceToDevice, stream);
    k_gated_wmma<<<ggrid,64,0,stream>>>(cur, edgeF, nxt, gbuf, edge_src, edge_dst,
        nuW0h + (size_t)(i*2+0)*192*64, nuW1h + (size_t)(i*2+0)*64*64,
        nuW2h + (size_t)(i*2+0)*64*64,
        nu_b0 + (i*2+0)*64, nu_b1 + (i*2+0)*64, nu_b2 + (i*2+0)*64,
        nw_W + (size_t)i*9*64, rbf, E, 2);
    float* t2 = cur; cur = nxt; nxt = t2;
  }

  k_gather<<<(B*64+255)/256,256,0,stream>>>(cur, absorber, (float*)d_out, B*64);
}
